// SumTTLinear_1125281431958
// MI455X (gfx1250) — compile-verified, matrix operand order checked by reading, not run on hardware
//
#include <hip/hip_runtime.h>
#include <hip/hip_bf16.h>
#include <stdint.h>

// ---------------------------------------------------------------------------
// SumTTLinear on MI455X (gfx1250):
//   kernel 1: contract D_k x U_k -> joint cores, stored as pre-swizzled bf16
//             WMMA A-fragments (ISA 7.12.2 16-bit A layout).
//             K-axis order: stage 0 -> K=i ; stages 1..3 -> K = i*16 + g
//             (so the stage shuffle can write one contiguous b128 per lane).
//   kernel 2: one workgroup per token; 4 chained GEMM stages, all operands in
//             LDS (two ping-pong 128x512 bf16 buffers, K padded to 136 for
//             bank-conflict-free b128 column reads). Two-accumulator software
//             pipeline with full-tile load->use distance for both buffers.
//             Z0 lives permanently in the 8 pad rows (128..135) of bufA.
// ---------------------------------------------------------------------------

typedef __bf16 v16bf __attribute__((ext_vector_type(16)));
typedef float  v8f   __attribute__((ext_vector_type(8)));

#define TT_ITERS   5
#define TT_PERITER 38912   // bf16 A-fragment elements per iteration
#define KCAP       136u    // padded K rows per LDS column (128 + 8 = Z0 home)

__device__ __forceinline__ uint16_t f2bf(float f) {
  union { float f; uint32_t u; } c; c.f = f;
  uint32_t u = c.u + 0x7FFFu + ((c.u >> 16) & 1u);   // round-to-nearest-even
  return (uint16_t)(u >> 16);
}

// Pack two f32 into one dword of bf16 (a -> [15:0], b -> [31:16]).
// Fast path: hardware pack if available; else round-nearest(+0x8000) and one
// v_perm_b32 splicing the two high halves -> 3 VALU per pair.
__device__ __forceinline__ uint32_t pack_bf16(float a, float b) {
#if __has_builtin(__builtin_amdgcn_cvt_pk_bf16_f32)
  return __builtin_bit_cast(uint32_t, __builtin_amdgcn_cvt_pk_bf16_f32(a, b));
#else
  uint32_t ra = __builtin_bit_cast(uint32_t, a) + 0x8000u;
  uint32_t rb = __builtin_bit_cast(uint32_t, b) + 0x8000u;
  return __builtin_amdgcn_perm(rb, ra, 0x07060302u);  // {rb.hi16, ra.hi16}
#endif
}

// K index held by (element e, lane-half h) in the 16-bit A/B fragment layout.
__device__ __forceinline__ int kmap(int e, int h) {
  int v = e >> 1, lo = e & 1;
  int base = (v < 4) ? (2 * v) : (16 + 2 * (v - 4));
  return base + lo + 8 * h;
}

// ---------------------------------------------------------------------------
// Kernel 1: build joint cores, pre-swizzled into A-fragment order.
// Layout: aswz[it][stage][(tr*KB+kb)][lane][e]  (bf16)
// Stage element counts: s0=4096 (K padded 8->32), s1=s2=16384, s3=2048 (M 8->16)
// ---------------------------------------------------------------------------
__global__ void tt_build_cores(const float* __restrict__ d0, const float* __restrict__ d1,
                               const float* __restrict__ d2, const float* __restrict__ d3,
                               const float* __restrict__ u0, const float* __restrict__ u1,
                               const float* __restrict__ u2, const float* __restrict__ u3,
                               uint16_t* __restrict__ aswz, int total) {
  int gid = blockIdx.x * blockDim.x + threadIdx.x;
  if (gid >= total) return;

  int it  = gid / TT_PERITER;
  int rem = gid - it * TT_PERITER;
  int s, off;
  if      (rem < 4096)  { s = 0; off = rem; }
  else if (rem < 20480) { s = 1; off = rem - 4096; }
  else if (rem < 36864) { s = 2; off = rem - 20480; }
  else                  { s = 3; off = rem - 36864; }

  const int KBs[4]  = {1, 4, 4, 4};   // K blocks of 32
  const int Ssz[4]  = {1, 4, 4, 4};   // U leading rank
  const int Rsz[4]  = {1, 4, 4, 4};   // D leading rank
  const int Asz[4]  = {4, 4, 4, 1};   // D trailing rank
  const int Bsz[4]  = {4, 4, 4, 1};   // U trailing rank
  const int Dstr[4] = {512, 2048, 2048, 512};  // per-iter strides
  const int Ustr[4] = {512, 2048, 2048, 512};

  int e    = off & 15;
  int lane = (off >> 4) & 31;
  int tkb  = off >> 9;
  int kb   = tkb % KBs[s];
  int tr   = tkb / KBs[s];
  int h    = lane >> 4;
  int m    = tr * 16 + (lane & 15);         // output row (o,g)
  int K    = kb * 32 + kmap(e, h);          // contraction index

  int o, g;
  bool valid = true;
  if (s == 3) { o = m; g = 0; if (m >= 8) valid = false; }
  else        { o = m >> 4; g = m & 15; }

  // K-axis decode: stage 0 uses K = i (rows 8..31 zero); stages 1..3 use
  // K = i*16 + rank  (matches the contiguous-b128 shuffle in kernel 2).
  int i, rs;
  if (s == 0) { i = K; rs = 0; if (K >= 8) valid = false; }
  else        { i = K >> 4; rs = K & 15; }

  int r  = rs / Ssz[s], si = rs % Ssz[s];
  if (r >= Rsz[s]) valid = false;
  int a = g / Bsz[s], b = g % Bsz[s];

  float acc = 0.f;
  if (valid) {
    const float* D = (s == 0) ? d0 : (s == 1) ? d1 : (s == 2) ? d2 : d3;
    const float* U = (s == 0) ? u0 : (s == 1) ? u1 : (s == 2) ? u2 : u3;
    const int dA = Asz[s], uB = Bsz[s];
    const float* Dp = D + (size_t)it * Dstr[s] + (size_t)((r * 8 + i) * 16) * dA + a;
    const float* Up = U + (size_t)it * Ustr[s] + (size_t)(si * 16 * 8 + o) * uB + b;
#pragma unroll
    for (int mm = 0; mm < 16; ++mm)
      acc += Dp[mm * dA] * Up[mm * 8 * uB];
  }
  aswz[gid] = f2bf(acc);
}

// ---------------------------------------------------------------------------
// Kernel 2 helpers
// ---------------------------------------------------------------------------
union AFrag { v16bf v; uint4 q[2]; };

template <int S>
__device__ __forceinline__ void tt_stage(const uint16_t* __restrict__ abase,
                                         const uint16_t* __restrict__ src,
                                         uint16_t* __restrict__ dst,
                                         float* __restrict__ accF,
                                         const uint16_t* __restrict__ zlds,
                                         unsigned wave, unsigned lane,
                                         unsigned h, unsigned ln) {
  constexpr unsigned KB     = (S == 0) ? 1 : 4;   // K blocks of 32
  constexpr unsigned NTILES = (S == 3) ? 4 : 32;  // column tiles per wave
  const unsigned tr = (S == 3) ? 0u : wave;       // tile row is wave-invariant

  // This wave's A strip (16 x 32*KB), loaded once: coalesced 32B per lane.
  AFrag af[KB];
#pragma unroll
  for (unsigned kb = 0; kb < KB; ++kb) {
    const uint4* ap = (const uint4*)(abase + (size_t)((tr * KB + kb) * 32 + lane) * 16);
    af[kb].q[0] = ap[0];
    af[kb].q[1] = ap[1];
  }

  auto tcOf = [&](unsigned idx) { return (S == 3) ? (wave + idx * 8) : idx; };

  // B fragment loads: two 16B-aligned ds_load_b128 per K-block (stride-padded
  // so 16 columns fan across all 64 banks). Stage 0: only K rows 0..7 are
  // nonzero -> per-lane ADDRESS select between Z0 pad rows and a zero strip.
  auto loadB = [&](unsigned tc, AFrag* b) {
    if (S == 0) {
      const uint16_t* a0 =
          (h == 0) ? (src + (size_t)(tc * 16 + ln) * KCAP + 128) : zlds;
      b[0].q[0] = *(const uint4*)a0;
      b[0].q[1] = *(const uint4*)zlds;
    } else {
#pragma unroll
      for (unsigned kb = 0; kb < KB; ++kb) {
        const uint16_t* colp = src + (size_t)(tc * 16 + ln) * KCAP + kb * 32 + 8 * h;
        b[kb].q[0] = *(const uint4*)(colp);
        b[kb].q[1] = *(const uint4*)(colp + 16);
      }
    }
  };

  auto chain = [&](AFrag* b) -> v8f {
    v8f c;
#pragma unroll
    for (int j = 0; j < 8; ++j) c[j] = 0.f;
#pragma unroll
    for (unsigned kb = 0; kb < KB; ++kb)
      c = __builtin_amdgcn_wmma_f32_16x16x32_bf16(
          false, af[kb].v, false, b[kb].v, (short)0, c, false, false);
    return c;
  };

  auto storeC = [&](v8f& c, unsigned tc) {
    const unsigned col = tc * 16 + ln;
    if (S < 3) {
      // C[(o,g),(prev,i_next,f')] -> Z'[K'=i_next*16+g][(prev*8+o, f')]
      // o == tr for all 8 rows; zrow = inext*16 + 8h + j -> the lane's 8
      // outputs are CONTIGUOUS: pack to 4 dwords, one 16B ds_store_b128.
      constexpr unsigned fcur  = 512u >> (3 * S);
      constexpr unsigned fnext = fcur >> 3;
      const unsigned prev  = col / fcur;
      const unsigned fidx  = col % fcur;
      const unsigned inext = fidx / fnext;
      const unsigned fpi   = fidx % fnext;
      const unsigned zcol  = (prev * 8 + tr) * fnext + fpi;
      uint16_t* p = dst + (size_t)zcol * KCAP + inext * 16 + 8 * h;
      uint4 q;
      q.x = pack_bf16(c[0], c[1]);
      q.y = pack_bf16(c[2], c[3]);
      q.z = pack_bf16(c[4], c[5]);
      q.w = pack_bf16(c[6], c[7]);
      *(uint4*)p = q;
    } else {
      // Stage 3: rows 0..7 are o4 (h==1 rows are zero padding).
      if (h == 0) {
        float* p = accF + col * 8;
#pragma unroll
        for (int j = 0; j < 8; ++j) atomicAdd(p + j, c[j]);  // ds_add_f32
      }
    }
  };

  // Two-accumulator software pipeline; each buffer's loads are issued one
  // full tile (chain + store) before consumption, so the scheduler can use
  // partial s_wait_dscnt and overlap LDS latency with WMMA + pack VALU.
  AFrag b0[KB], b1[KB];
  v8f c0, c1;
  loadB(tcOf(0), b0);
  loadB(tcOf(1), b1);
  c0 = chain(b0);
  for (unsigned idx = 0; idx + 2 < NTILES; idx += 2) {
    loadB(tcOf(idx + 2), b0);
    c1 = chain(b1);
    storeC(c0, tcOf(idx));
    loadB(tcOf(idx + 3), b1);
    c0 = chain(b0);
    storeC(c1, tcOf(idx + 1));
  }
  c1 = chain(b1);
  storeC(c0, tcOf(NTILES - 2));
  storeC(c1, tcOf(NTILES - 1));
}

// ---------------------------------------------------------------------------
// Kernel 2: one block per token, 256 threads (8 wave32), all-WMMA pipeline.
// LDS: bufA/bufB = 512 cols x KCAP bf16 (139264 B each), accF = 4096 f32,
//      16B zero strip for stage-0 operand padding.
// ---------------------------------------------------------------------------
__global__ __launch_bounds__(256) void tt_gemm_wmma(
    const float* __restrict__ x, const uint16_t* __restrict__ aswz,
    const float* __restrict__ bias, float* __restrict__ out) {
  extern __shared__ __align__(16) char smem[];
  uint16_t* bufA = (uint16_t*)smem;                          // ping (+ Z0 home)
  uint16_t* bufB = (uint16_t*)(smem + 512 * KCAP * 2);       // pong
  float*    accF = (float*)(smem + 2 * (512 * KCAP * 2));    // 4096 f32
  uint16_t* zlds = (uint16_t*)(smem + 2 * (512 * KCAP * 2) + 4096 * 4); // 16B zeros

  const unsigned tid  = threadIdx.x;
  const unsigned wave = tid >> 5;
  const unsigned lane = tid & 31;
  const unsigned h    = lane >> 4;
  const unsigned ln   = lane & 15;
  const unsigned n    = blockIdx.x;
  const float* xr = x + (size_t)n * 4096;

  // Persistent Z0: rows 128..135 of bufA (stage 1 only writes rows 0..127,
  // so this survives all 5 iterations). Filled once, coalesced.
  for (unsigned j = tid; j < 4096; j += 256) {
    unsigned i1 = j >> 9, f = j & 511;
    bufA[f * KCAP + 128 + i1] = f2bf(xr[j]);
  }
  for (unsigned j = tid; j < 4096; j += 256) accF[j] = 0.f;
  if (tid == 0) {
    uint4 zq; zq.x = 0; zq.y = 0; zq.z = 0; zq.w = 0;
    *(uint4*)zlds = zq;
  }
  __syncthreads();

  for (int it = 0; it < TT_ITERS; ++it) {
    const uint16_t* ab = aswz + (size_t)it * TT_PERITER;
    tt_stage<0>(ab,         bufA, bufB, accF, zlds, wave, lane, h, ln);
    __syncthreads();
    tt_stage<1>(ab + 4096,  bufB, bufA, accF, zlds, wave, lane, h, ln);
    __syncthreads();
    tt_stage<2>(ab + 20480, bufA, bufB, accF, zlds, wave, lane, h, ln);
    __syncthreads();
    tt_stage<3>(ab + 36864, bufB, bufA, accF, zlds, wave, lane, h, ln);
    __syncthreads();
  }

  for (unsigned j = tid; j < 4096; j += 256)
    out[(size_t)n * 4096 + j] = accF[j] + bias[j];
}

// ---------------------------------------------------------------------------
extern "C" void kernel_launch(void* const* d_in, const int* in_sizes, int n_in,
                              void* d_out, int out_size, void* d_ws, size_t ws_size,
                              hipStream_t stream) {
  (void)in_sizes; (void)n_in; (void)out_size; (void)ws_size;
  const float* x    = (const float*)d_in[0];
  const float* d0   = (const float*)d_in[1];
  const float* d1   = (const float*)d_in[2];
  const float* d2   = (const float*)d_in[3];
  const float* d3   = (const float*)d_in[4];
  const float* u0   = (const float*)d_in[5];
  const float* u1   = (const float*)d_in[6];
  const float* u2   = (const float*)d_in[7];
  const float* u3   = (const float*)d_in[8];
  const float* bias = (const float*)d_in[9];
  float* out = (float*)d_out;

  uint16_t* aswz = (uint16_t*)d_ws;   // 194560 bf16 = 389120 B of scratch
  const int total = TT_ITERS * TT_PERITER;

  tt_build_cores<<<(total + 255) / 256, 256, 0, stream>>>(
      d0, d1, d2, d3, u0, u1, u2, u3, aswz, total);

  const size_t smem = (size_t)2 * (512 * KCAP * 2) + 4096 * 4 + 16; // 294928 B
  (void)hipFuncSetAttribute(reinterpret_cast<const void*>(tt_gemm_wmma),
                            hipFuncAttributeMaxDynamicSharedMemorySize, (int)smem);
  tt_gemm_wmma<<<512, 256, smem, stream>>>(x, aswz, bias, out);
}
